// WaveNet_25494925869404
// MI455X (gfx1250) — compile-verified, matrix operand order checked by reading, not run on hardware
//
#include <hip/hip_runtime.h>
#include <hip/hip_bf16.h>
#include <math.h>

// ---------------------------------------------------------------------------
// WaveNet forward for MI455X (gfx1250, wave32, WMMA).
// All GEMMs use v_wmma_f32_16x16x32_f16 (f16 in, f32 accumulate).
// Weight panels DMA'd to LDS via global_load_async_to_lds_b128 (ASYNCcnt);
// 4 N-tiles per wave -> 4 independent WMMA chains; B operands software-
// pipelined one K-step ahead.
// ---------------------------------------------------------------------------

typedef __attribute__((ext_vector_type(16))) _Float16 v16h;
typedef __attribute__((ext_vector_type(8)))  _Float16 v8h;
typedef __attribute__((ext_vector_type(8)))  float    v8f;

#define WMMA16(a, b, c) __builtin_amdgcn_wmma_f32_16x16x32_f16( \
    false, (a), false, (b), (short)0, (c), false, false)

__device__ __forceinline__ v8f vzero8f() {
  v8f z;
#pragma unroll
  for (int i = 0; i < 8; ++i) z[i] = 0.0f;
  return z;
}

// Async DMA one 16-byte chunk global -> LDS (CDNA5 GLOBAL_LOAD_ASYNC_TO_LDS).
__device__ __forceinline__ void async_g2l_b128(void* lds, const void* gaddr) {
  unsigned loff = (unsigned)(reinterpret_cast<size_t>(lds));  // LDS byte offset
  asm volatile("global_load_async_to_lds_b128 %0, %1, off"
               :: "v"(loff), "v"(gaddr) : "memory");
}
__device__ __forceinline__ void wait_async_zero() {
  asm volatile("s_wait_asynccnt 0x0" ::: "memory");
}

// B operand (K=32 x N=16 tile): lane group lo needs k{0..7,16..23}, hi needs
// k{8..15,24..31}. With channel-fast rows that's two contiguous b128 loads.
__device__ __forceinline__ v16h load_b_tile(const _Float16* __restrict__ row, int kb) {
  v8h lo = *(const v8h*)(row + kb);
  v8h hi = *(const v8h*)(row + kb + 16);
  v16h r;
#pragma unroll
  for (int j = 0; j < 8; ++j) { r[j] = lo[j]; r[j + 8] = hi[j]; }
  return r;
}

__device__ __forceinline__ float elu_f(float x) {
  return x > 0.0f ? x : (__expf(x) - 1.0f);
}

// ---------------------------------------------------------------------------
// Weight packer: emit per-lane WMMA A-layout tiles (16x32 f16 per tile).
// dst[(mt*Kt+kt)*512 + lane*16 + j]; src index = base + layer*srcLs
//   + m*sm + (k/32)*sko + (k%32)*ski   (covers [O][C], [O][C][tap], [L][O][C]).
// ---------------------------------------------------------------------------
__global__ void k_pack(const float* __restrict__ src, _Float16* __restrict__ dst,
                       int Mt, int Kt, int sm, int sko, int ski, int base,
                       long srcLs, long dstLs) {
  int total = Mt * Kt * 512;
  int idx = blockIdx.x * blockDim.x + threadIdx.x;
  if (idx >= total) return;
  int layer = blockIdx.y;
  int tile = idx >> 9, rem = idx & 511;
  int lane = rem >> 4, j = rem & 15;
  int mt = tile / Kt, kt = tile % Kt;
  int m  = mt * 16 + (lane & 15);
  int kb = (lane < 16) ? 0 : 8;
  int k  = kt * 32 + kb + j + ((j >= 8) ? 8 : 0);
  long s = (long)base + (long)layer * srcLs + (long)m * sm
         + (long)(k >> 5) * sko + (long)(k & 31) * ski;
  dst[(long)layer * dstLs + idx] = (_Float16)src[s];
}

// x [B,256,L] f32 (NCL) -> xh [B][L][256] f16 (channel-fast)
__global__ void k_xpose(const float* __restrict__ x, _Float16* __restrict__ xh) {
  long idx = (long)blockIdx.x * blockDim.x + threadIdx.x;
  const long N = 8L * 16384 * 256;
  if (idx >= N) return;
  long b = idx / (16384L * 256), rem = idx % (16384L * 256);
  long p = rem / 256, c = rem % 256;
  xh[idx] = (_Float16)x[(b * 256 + c) * 16384 + p];
}

// bias for the fused skip GEMM = sum over the 50 layers' skip biases
__global__ void k_bias_sum(const float* __restrict__ bskip, float* __restrict__ out) {
  int m = blockIdx.x * blockDim.x + threadIdx.x;
  if (m >= 512) return;
  float s = 0.0f;
  for (int i = 0; i < 50; ++i) s += bskip[i * 512 + m];
  out[m] = s;
}

// ---------------------------------------------------------------------------
// Generic WMMA GEMM: Y[r, m] = sum_k A[m,k] * B[r,k] + bias[m]
// Block = 8 waves, all sharing one mt; A panel (Kt x 1KB) async-DMA'd to LDS.
// Each wave owns 4 consecutive 16-row N-tiles -> 4 independent WMMA chains,
// with B operands pipelined one K-step ahead.
// mode 0: store f32 [r][ldo] and f16 [r][ldo]   (input conv -> residual stream)
// mode 1: ELU then f16 [r][ldo]                 (skip-fused / post1)
// mode 2: f32 NCH output d_out[b][m][p]         (post2, finalLen decodes b,p)
// ---------------------------------------------------------------------------
__global__ void k_gemm(const _Float16* __restrict__ Apk,
                       const _Float16* __restrict__ Bh, int ldb,
                       const float* __restrict__ bias,
                       int Mt, int Kt, long R, int mode,
                       float* __restrict__ outF, _Float16* __restrict__ outH,
                       int ldo, int finalLen) {
  extern __shared__ __align__(32) _Float16 As[];   // Kt * 512 halves
  int mt = (int)(blockIdx.x % (unsigned)Mt);
  long ng = (long)(blockIdx.x / (unsigned)Mt);
  int lane = threadIdx.x & 31;
  int w    = threadIdx.x >> 5;

  // Async DMA the A panel into LDS (no VGPR round-trip).
  {
    const v8h* gsrc = (const v8h*)(Apk + (long)mt * Kt * 512);
    v8h* ldst = (v8h*)As;
    for (int c = threadIdx.x; c < Kt * 64; c += blockDim.x)
      async_g2l_b128(ldst + c, gsrc + c);
    wait_async_zero();
  }
  __syncthreads();

  long r0w = ng * 512 + (long)w * 64;              // this wave's 4x16 rows
  if (r0w >= R) return;                            // wave-uniform

  int nlane = lane & 15, hi4 = lane >> 4, kb = hi4 * 8;
  const _Float16* brow[4];
  long rowv[4];
#pragma unroll
  for (int j = 0; j < 4; ++j) {
    long row = r0w + j * 16 + nlane;
    rowv[j] = row;
    long rowc = row < R ? row : R - 1;
    brow[j] = Bh + rowc * (long)ldb;
  }

  v8f acc[4];
  v16h bcur[4];
#pragma unroll
  for (int j = 0; j < 4; ++j) {
    acc[j] = vzero8f();
    bcur[j] = load_b_tile(brow[j], kb);
  }

  for (int kt = 0; kt < Kt; ++kt) {
    v16h a = *(const v16h*)(As + kt * 512 + lane * 16);   // ds_load
    v16h bnxt[4];
#pragma unroll
    for (int j = 0; j < 4; ++j) bnxt[j] = bcur[j];
    if (kt + 1 < Kt) {
#pragma unroll
      for (int j = 0; j < 4; ++j)
        bnxt[j] = load_b_tile(brow[j] + (kt + 1) * 32, kb);  // prefetch K+1
    }
#pragma unroll
    for (int j = 0; j < 4; ++j)
      acc[j] = WMMA16(a, bcur[j], acc[j]);           // 4 independent chains
#pragma unroll
    for (int j = 0; j < 4; ++j) bcur[j] = bnxt[j];
  }

  int mbase = mt * 16 + kb;
#pragma unroll
  for (int j = 0; j < 4; ++j) {
    long row = rowv[j];
    bool valid = row < R;
    if (mode == 0) {
      v8f of; v8h oh;
#pragma unroll
      for (int r = 0; r < 8; ++r) {
        float v = acc[j][r] + bias[mbase + r];
        of[r] = v; oh[r] = (_Float16)v;
      }
      if (valid) {
        *(v8f*)(outF + row * (long)ldo + mbase) = of;
        *(v8h*)(outH + row * (long)ldo + mbase) = oh;
      }
    } else if (mode == 1) {
      v8h oh;
#pragma unroll
      for (int r = 0; r < 8; ++r) {
        float v = elu_f(acc[j][r] + bias[mbase + r]);
        oh[r] = (_Float16)v;
      }
      if (valid) *(v8h*)(outH + row * (long)ldo + mbase) = oh;
    } else {
      if (valid) {
        long b = row / finalLen, p = row % finalLen;
#pragma unroll
        for (int r = 0; r < 8; ++r)
          outF[(b * (long)ldo + (mbase + r)) * (long)finalLen + p] =
              acc[j][r] + bias[mbase + r];
      }
    }
  }
}

// ---------------------------------------------------------------------------
// One WaveNet layer. Absolute positions: out_new[p] = res[p] + out_old[p];
// gate[p] = sigmoid(Wsig0 x[p-d] + Wsig1 x[p] + bs) * tanh(same w/ Wtanh).
// All 10 weight tiles (4 sig + 4 tanh + 2 res, 10 KB) async-DMA'd to LDS.
// Gate D-tiles feed the K=32 residual WMMA as B operand with NO shuffles
// (D layout {n=l&15, m=r+8*hi} == B layout {n=l&15, k in kb+{0..7,16..23}}).
// ---------------------------------------------------------------------------
__global__ void k_layer(const _Float16* __restrict__ xh, const float* __restrict__ xf,
                        _Float16* __restrict__ yh, float* __restrict__ yf,
                        _Float16* __restrict__ gates,
                        const _Float16* __restrict__ wsig,
                        const _Float16* __restrict__ wtanh,
                        const _Float16* __restrict__ wres,
                        const float* __restrict__ bsig,
                        const float* __restrict__ btanh,
                        const float* __restrict__ bres,
                        int d, int pstart, int layer) {
  const int L = 16384, FIN = 11269, TRIM = 5115;
  __shared__ __align__(32) _Float16 Wlds[5120];  // [0,2048)=sig [2048,4096)=tanh [4096,5120)=res

  {
    v8h* l8 = (v8h*)Wlds;
    for (int c = threadIdx.x; c < 640; c += blockDim.x) {
      const v8h* src;
      if (c < 256)      src = (const v8h*)wsig  + c;
      else if (c < 512) src = (const v8h*)wtanh + (c - 256);
      else              src = (const v8h*)wres  + (c - 512);
      async_g2l_b128(l8 + c, src);
    }
    wait_async_zero();
  }
  __syncthreads();

  long wave = (long)blockIdx.x * (blockDim.x >> 5) + (threadIdx.x >> 5);
  int  lane = threadIdx.x & 31;
  int  ntiles = (L - pstart + 15) >> 4;
  if (wave >= 8L * ntiles) return;                 // wave-uniform
  int b = (int)(wave / ntiles), t = (int)(wave % ntiles);
  int nlane = lane & 15, hi4 = lane >> 4, kb = hi4 * 8;
  int p  = pstart + t * 16 + nlane;
  int pc = p < L ? p : L - 1;
  bool valid = p < L;

  const _Float16* rowhi = xh + ((long)b * L + pc) * 32;
  const _Float16* rowlo = rowhi - (long)d * 32;    // p-d >= pstart-d >= 0
  v16h Blo = load_b_tile(rowlo, kb);
  v16h Bhi = load_b_tile(rowhi, kb);

  float g[2][8];
#pragma unroll
  for (int mt = 0; mt < 2; ++mt) {
    v8f aS = vzero8f(), aT = vzero8f();
    v16h s0 = *(const v16h*)(Wlds +        (0 * 2 + mt) * 512 + lane * 16);
    v16h s1 = *(const v16h*)(Wlds +        (1 * 2 + mt) * 512 + lane * 16);
    v16h t0 = *(const v16h*)(Wlds + 2048 + (0 * 2 + mt) * 512 + lane * 16);
    v16h t1 = *(const v16h*)(Wlds + 2048 + (1 * 2 + mt) * 512 + lane * 16);
    aS = WMMA16(s0, Blo, aS);
    aT = WMMA16(t0, Blo, aT);
    aS = WMMA16(s1, Bhi, aS);
    aT = WMMA16(t1, Bhi, aT);
    int mbase = mt * 16 + kb;
#pragma unroll
    for (int r = 0; r < 8; ++r) {
      float sv = aS[r] + bsig[mbase + r];
      float tv = aT[r] + btanh[mbase + r];
      g[mt][r] = (1.0f / (1.0f + __expf(-sv))) * tanhf(tv);
    }
  }

  // Persist gate f16 into the fused-skip buffer [b][p-TRIM][layer*32 + c]
  if (valid && p >= TRIM) {
    _Float16* grow = gates + ((long)b * FIN + (p - TRIM)) * 1600 + layer * 32;
    v8h g0, g1;
#pragma unroll
    for (int r = 0; r < 8; ++r) { g0[r] = (_Float16)g[0][r]; g1[r] = (_Float16)g[1][r]; }
    *(v8h*)(grow + kb)      = g0;   // channels kb..kb+7
    *(v8h*)(grow + 16 + kb) = g1;   // channels 16+kb..16+kb+7
  }

  // Gate as WMMA B operand, directly from the D-tile registers.
  v16h Bg;
#pragma unroll
  for (int j = 0; j < 8; ++j) { Bg[j] = (_Float16)g[0][j]; Bg[j + 8] = (_Float16)g[1][j]; }

  // Residual: out_new = Wres @ gate + bres + out_old   (ping -> pong)
#pragma unroll
  for (int mt = 0; mt < 2; ++mt) {
    v8f acc = vzero8f();
    v16h a = *(const v16h*)(Wlds + 4096 + mt * 512 + lane * 16);
    acc = WMMA16(a, Bg, acc);
    int mbase = mt * 16 + kb;
    v8f oldv = *(const v8f*)(xf + ((long)b * L + pc) * 32 + mbase);
    v8f nf; v8h nh;
#pragma unroll
    for (int r = 0; r < 8; ++r) {
      float v = acc[r] + bres[mbase + r] + oldv[r];
      nf[r] = v; nh[r] = (_Float16)v;
    }
    if (valid) {
      *(v8f*)(yf + ((long)b * L + p) * 32 + mbase) = nf;
      *(v8h*)(yh + ((long)b * L + p) * 32 + mbase) = nh;
    }
  }
}

// ---------------------------------------------------------------------------
extern "C" void kernel_launch(void* const* d_in, const int* in_sizes, int n_in,
                              void* d_out, int out_size, void* d_ws, size_t ws_size,
                              hipStream_t stream) {
  const long B = 8, L = 16384, FIN = 11269;
  const float* x       = (const float*)d_in[0];
  const float* w_in    = (const float*)d_in[1];
  const float* b_in    = (const float*)d_in[2];
  const float* w_sig   = (const float*)d_in[3];
  const float* b_sig   = (const float*)d_in[4];
  const float* w_tanh  = (const float*)d_in[5];
  const float* b_tanh  = (const float*)d_in[6];
  const float* w_skip  = (const float*)d_in[7];
  const float* b_skip  = (const float*)d_in[8];
  const float* w_res   = (const float*)d_in[9];
  const float* b_res   = (const float*)d_in[10];
  const float* w_post1 = (const float*)d_in[11];
  const float* b_post1 = (const float*)d_in[12];
  const float* w_post2 = (const float*)d_in[13];
  const float* b_post2 = (const float*)d_in[14];

  char* ws = (char*)d_ws;
  size_t off = 0;
  auto alloc = [&](size_t bytes) -> void* {
    void* p = (void*)(ws + off);
    off = (off + bytes + 255) & ~(size_t)255;
    return p;
  };
  _Float16* xh    = (_Float16*)alloc(B * L * 256 * 2);
  float*    resF0 = (float*)   alloc(B * L * 32 * 4);
  float*    resF1 = (float*)   alloc(B * L * 32 * 4);
  _Float16* resH0 = (_Float16*)alloc(B * L * 32 * 2);
  _Float16* resH1 = (_Float16*)alloc(B * L * 32 * 2);
  _Float16* gates = (_Float16*)alloc(B * FIN * 1600 * 2);
  _Float16* h1    = (_Float16*)alloc(B * FIN * 512 * 2);
  _Float16* h2    = (_Float16*)alloc(B * FIN * 512 * 2);
  _Float16* wpin  = (_Float16*)alloc(2L * 8 * 512 * 2);
  _Float16* wpsig = (_Float16*)alloc(50L * 4 * 512 * 2);
  _Float16* wptan = (_Float16*)alloc(50L * 4 * 512 * 2);
  _Float16* wpres = (_Float16*)alloc(50L * 2 * 512 * 2);
  _Float16* wpskp = (_Float16*)alloc(32L * 50 * 512 * 2);
  _Float16* wpp1  = (_Float16*)alloc(32L * 16 * 512 * 2);
  _Float16* wpp2  = (_Float16*)alloc(16L * 16 * 512 * 2);
  float*    bskps = (float*)   alloc(512 * 4);

  // --- weight packing into per-lane WMMA A-layout -------------------------
  k_pack<<<dim3((2 * 8 * 512 + 255) / 256, 1), 256, 0, stream>>>(
      w_in, wpin, 2, 8, 256, 32, 1, 0, 0, 0);
  k_pack<<<dim3(4, 50), 256, 0, stream>>>(w_sig,  wpsig,        2, 1, 64, 64, 2, 0, 2048, 2048);
  k_pack<<<dim3(4, 50), 256, 0, stream>>>(w_sig,  wpsig + 1024, 2, 1, 64, 64, 2, 1, 2048, 2048);
  k_pack<<<dim3(4, 50), 256, 0, stream>>>(w_tanh, wptan,        2, 1, 64, 64, 2, 0, 2048, 2048);
  k_pack<<<dim3(4, 50), 256, 0, stream>>>(w_tanh, wptan + 1024, 2, 1, 64, 64, 2, 1, 2048, 2048);
  k_pack<<<dim3(4, 50), 256, 0, stream>>>(w_res, wpres, 2, 1, 32, 1024, 1, 0, 1024, 1024);
  k_pack<<<dim3((32 * 50 * 512 + 255) / 256, 1), 256, 0, stream>>>(
      w_skip, wpskp, 32, 50, 32, 16384, 1, 0, 0, 0);
  k_pack<<<dim3((32 * 16 * 512 + 255) / 256, 1), 256, 0, stream>>>(
      w_post1, wpp1, 32, 16, 512, 32, 1, 0, 0, 0);
  k_pack<<<dim3((16 * 16 * 512 + 255) / 256, 1), 256, 0, stream>>>(
      w_post2, wpp2, 16, 16, 512, 32, 1, 0, 0, 0);
  k_bias_sum<<<2, 256, 0, stream>>>(b_skip, bskps);

  // --- x NCL f32 -> [b][p][c] f16 -----------------------------------------
  long Nx = B * L * 256;
  k_xpose<<<(unsigned)((Nx + 255) / 256), 256, 0, stream>>>(x, xh);

  // --- input 1x1 conv (K=256 GEMM) -> residual stream ping ----------------
  long R0 = B * L;
  {
    long ngroups = (R0 + 511) / 512;
    k_gemm<<<(unsigned)(2 * ngroups), 256, 8 * 1024, stream>>>(
        wpin, xh, 256, b_in, 2, 8, R0, 0, resF0, resH0, 32, 0);
  }

  // --- 50 gated residual layers, ping-pong --------------------------------
  _Float16 *curH = resH0, *nxtH = resH1;
  float    *curF = resF0, *nxtF = resF1;
  int s = 0;
  for (int i = 0; i < 50; ++i) {
    int d = 1 << (i % 10);
    int pstart = s + d;                       // output valid for p in [pstart, L)
    long ntiles = (L - pstart + 15) / 16;
    long waves = 8 * ntiles;
    k_layer<<<(unsigned)((waves + 7) / 8), 256, 0, stream>>>(
        curH, curF, nxtH, nxtF, gates,
        wpsig + (long)i * 2048, wptan + (long)i * 2048, wpres + (long)i * 1024,
        b_sig + i * 32, b_tanh + i * 32, b_res + i * 32, d, pstart, i);
    _Float16* th = curH; curH = nxtH; nxtH = th;
    float*    tf = curF; curF = nxtF; nxtF = tf;
    s = pstart;
  }

  // --- fused skip projection: one K=1600 GEMM, bias=sum(b_skip), ELU ------
  long R1 = B * FIN;
  long ng1 = (R1 + 511) / 512;
  k_gemm<<<(unsigned)(32 * ng1), 256, 50 * 1024, stream>>>(
      wpskp, gates, 1600, bskps, 32, 50, R1, 1, nullptr, h1, 512, 0);

  // --- post1 512->512 + ELU ------------------------------------------------
  k_gemm<<<(unsigned)(32 * ng1), 256, 16 * 1024, stream>>>(
      wpp1, h1, 512, b_post1, 32, 16, R1, 1, nullptr, h2, 512, 0);

  // --- post2 512->256 -> d_out [B][256][FIN] f32 --------------------------
  k_gemm<<<(unsigned)(16 * ng1), 256, 16 * 1024, stream>>>(
      wpp2, h2, 512, b_post2, 16, 16, R1, 2, (float*)d_out, nullptr, 256, FIN);
}
